// BFLinear_69604239999614
// MI455X (gfx1250) — matrix operand order repaired and checked by reading, not compiled, not run on hardware
//
#include <hip/hip_runtime.h>

typedef __attribute__((ext_vector_type(8)))  __bf16 v8bf;
typedef __attribute__((ext_vector_type(16))) __bf16 v16bf;
typedef __attribute__((ext_vector_type(8)))  float  v8f;

#define GEMM_M 8192
#define GEMM_N 4608
#define GEMM_K 4608
#define BM 256            // block tile M
#define BN 128            // block tile N
#define BK 32             // K-slab per LDS stage (one WMMA K step)
#define LDT 40            // padded LDS row stride in bf16 elems -> 80 bytes

// ---------------------------------------------------------------------------
// BFP quantize: groups of 36 contiguous elements share floor(log2(absmax)),
// mantissas truncated to 8 bits toward zero. Result is exactly representable
// in bf16 (<= 8 significant bits), so we emit raw bf16 bit patterns by
// truncating the f32 bits (low 16 bits are provably zero after quantization).
// ---------------------------------------------------------------------------
__global__ __launch_bounds__(256) void bfp_quant_kernel(const float* __restrict__ src,
                                                        unsigned short* __restrict__ dst,
                                                        long long nelem) {
  __shared__ float tile[9216];          // 256 groups * 36 elems
  __shared__ unsigned short sq[9216];
  const long long base = (long long)blockIdx.x * 9216;

  for (int i = threadIdx.x; i < 9216; i += 256) {
    long long idx = base + i;
    tile[i] = (idx < nelem) ? src[idx] : 0.0f;
  }
  __syncthreads();

  const float* gp = &tile[threadIdx.x * 36];
  float amax = 0.0f;
#pragma unroll
  for (int j = 0; j < 36; ++j) amax = fmaxf(amax, fabsf(gp[j]));

  unsigned short* qp = &sq[threadIdx.x * 36];
  if (amax > 0.0f) {
    unsigned int eb = (__float_as_uint(amax) >> 23) & 0xFFu;  // biased exponent
    if (eb < 8u) eb = 8u;                                     // denormal-group guard
    float step     = __uint_as_float((eb - 7u) << 23);        // 2^(e-7)
    float inv_step = __uint_as_float((261u - eb) << 23);      // 2^(7-e), exact
#pragma unroll
    for (int j = 0; j < 36; ++j) {
      float q = truncf(gp[j] * inv_step) * step;              // exact BFP value
      qp[j] = (unsigned short)(__float_as_uint(q) >> 16);     // exact -> bf16
    }
  } else {
#pragma unroll
    for (int j = 0; j < 36; ++j)
      qp[j] = (unsigned short)(__float_as_uint(gp[j]) >> 16);
  }
  __syncthreads();

  for (int i = threadIdx.x; i < 9216; i += 256) {
    long long idx = base + i;
    if (idx < nelem) dst[idx] = sq[i];
  }
}

// ---------------------------------------------------------------------------
// out[M,N] = qA[M,K](bf16) * qB[N,K](bf16)^T + bias, f32 accumulate via
// v_wmma_f32_16x16x32_bf16.
// Block tile 256x128, 8 waves, wave tile 64x64 (4x4 WMMA accumulators).
// Per stage per wave: 16 WMMA vs 16 ds_load_b128 (1.0 LDS reads / WMMA).
// Double-buffered LDS (60 KB) with register prefetch of the next K-slab.
// ---------------------------------------------------------------------------
__global__ __launch_bounds__(256) void bfp_gemm_wmma(const unsigned short* __restrict__ qA,
                                                     const unsigned short* __restrict__ qB,
                                                     const float* __restrict__ bias,
                                                     float* __restrict__ out) {
  __shared__ __align__(16) unsigned short As[2][BM * LDT];  // 2 x 20 KB
  __shared__ __align__(16) unsigned short Bs[2][BN * LDT];  // 2 x 10 KB

  const int tid  = threadIdx.x;
  const int lane = tid & 31;
  const int wave = tid >> 5;
  const int wm   = wave & 3;   // 4 waves along M, 64 rows each
  const int wn   = wave >> 2;  // 2 waves along N, 64 cols each
  const int lrow = lane & 15;
  const int lhi  = lane >> 4;  // lane-half select per WMMA operand layouts

  const int m0 = blockIdx.y * BM;
  const int n0 = blockIdx.x * BN;

  // Cooperative global->LDS mapping: 4 consecutive lanes cover 64 contiguous
  // bytes of one K-slab row (fully coalesced b128 loads).
  const int grow = tid >> 2;   // 0..63
  const int gseg = tid & 3;    // 4 x 16-byte segments per 32-elem row
  const unsigned short* gA = qA + (size_t)(m0 + grow) * GEMM_K + gseg * 8;
  const unsigned short* gB = qB + (size_t)(n0 + grow) * GEMM_K + gseg * 8;
  const int ldsoff = grow * LDT + gseg * 8;

  // Per-lane fragment offsets (ISA 16-bit A 16x32 / B 32x16 layouts).
  int aoff[4], boff[4];
#pragma unroll
  for (int mt = 0; mt < 4; ++mt)
    aoff[mt] = (wm * 64 + mt * 16 + lrow) * LDT + lhi * 8;   // + second chunk at +16
#pragma unroll
  for (int nt = 0; nt < 4; ++nt)
    boff[nt] = (wn * 64 + nt * 16 + lrow) * LDT + lhi * 8;   // + second chunk at +16

  v8f acc[4][4];
#pragma unroll
  for (int mt = 0; mt < 4; ++mt)
#pragma unroll
    for (int nt = 0; nt < 4; ++nt) {
      v8f z = {0.f, 0.f, 0.f, 0.f, 0.f, 0.f, 0.f, 0.f};
      acc[mt][nt] = z;
    }

  const int nk = GEMM_K / BK;  // 144

  // Prologue: fill buffer 0. (A: 4 row-blocks of 64; B: 2 row-blocks of 64.)
  v8bf ra[4], rb[2];
#pragma unroll
  for (int r = 0; r < 4; ++r) ra[r] = *(const v8bf*)(gA + (size_t)r * 64 * GEMM_K);
#pragma unroll
  for (int r = 0; r < 2; ++r) rb[r] = *(const v8bf*)(gB + (size_t)r * 64 * GEMM_K);
#pragma unroll
  for (int r = 0; r < 4; ++r) *(v8bf*)&As[0][ldsoff + r * 64 * LDT] = ra[r];
#pragma unroll
  for (int r = 0; r < 2; ++r) *(v8bf*)&Bs[0][ldsoff + r * 64 * LDT] = rb[r];
  __syncthreads();

  for (int kt = 0; kt < nk; ++kt) {
    const int cur = kt & 1;

    if (kt + 1 < nk) {  // prefetch next K-slab into registers
      const int kk = (kt + 1) * BK;
#pragma unroll
      for (int r = 0; r < 4; ++r)
        ra[r] = *(const v8bf*)(gA + (size_t)r * 64 * GEMM_K + kk);
#pragma unroll
      for (int r = 0; r < 2; ++r)
        rb[r] = *(const v8bf*)(gB + (size_t)r * 64 * GEMM_K + kk);
    }

    v16bf bfrag[4];
#pragma unroll
    for (int nt = 0; nt < 4; ++nt) {
      v8bf lo = *(const v8bf*)&Bs[cur][boff[nt]];
      v8bf hi = *(const v8bf*)&Bs[cur][boff[nt] + 16];
      bfrag[nt] = __builtin_shufflevector(lo, hi, 0, 1, 2, 3, 4, 5, 6, 7,
                                                  8, 9, 10, 11, 12, 13, 14, 15);
    }

#pragma unroll
    for (int mt = 0; mt < 4; ++mt) {
      v8bf lo = *(const v8bf*)&As[cur][aoff[mt]];
      v8bf hi = *(const v8bf*)&As[cur][aoff[mt] + 16];
      v16bf afrag = __builtin_shufflevector(lo, hi, 0, 1, 2, 3, 4, 5, 6, 7,
                                                    8, 9, 10, 11, 12, 13, 14, 15);
#pragma unroll
      for (int nt = 0; nt < 4; ++nt)
        acc[mt][nt] = __builtin_amdgcn_wmma_f32_16x16x32_bf16(
            false, afrag, false, bfrag[nt], (short)0, acc[mt][nt],
            false, false);
    }

    if (kt + 1 < nk) {
      const int nxt = (kt + 1) & 1;
      __syncthreads();  // prior reads of buffer `nxt` are done
#pragma unroll
      for (int r = 0; r < 4; ++r) *(v8bf*)&As[nxt][ldsoff + r * 64 * LDT] = ra[r];
#pragma unroll
      for (int r = 0; r < 2; ++r) *(v8bf*)&Bs[nxt][ldsoff + r * 64 * LDT] = rb[r];
      __syncthreads();  // stores visible before compute
    }
  }

  // Epilogue: C layout — VGPR v holds row (v + 8*lane_hi), col = lane%16.
#pragma unroll
  for (int mt = 0; mt < 4; ++mt) {
    const int row0 = m0 + wm * 64 + mt * 16 + lhi * 8;
#pragma unroll
    for (int nt = 0; nt < 4; ++nt) {
      const int col = n0 + wn * 64 + nt * 16 + lrow;
      const float bv = bias[col];
#pragma unroll
      for (int v = 0; v < 8; ++v)
        out[(size_t)(row0 + v) * GEMM_N + col] = acc[mt][nt][v] + bv;
    }
  }
}

// ---------------------------------------------------------------------------
extern "C" void kernel_launch(void* const* d_in, const int* in_sizes, int n_in,
                              void* d_out, int out_size, void* d_ws, size_t ws_size,
                              hipStream_t stream) {
  const float* inp  = (const float*)d_in[0];   // [8192, 4608]
  const float* wgt  = (const float*)d_in[1];   // [4608, 4608]
  const float* bias = (const float*)d_in[2];   // [4608]
  float* out = (float*)d_out;

  unsigned short* qx = (unsigned short*)d_ws;                 // 75.5 MB bf16
  unsigned short* qw = qx + (size_t)GEMM_M * GEMM_K;          // 42.5 MB bf16

  const long long nin = (long long)GEMM_M * GEMM_K;  // 37748736 = 4096*9216
  const long long nwt = (long long)GEMM_N * GEMM_K;  // 21233664 = 2304*9216
  bfp_quant_kernel<<<(int)(nin / 9216), 256, 0, stream>>>(inp, qx, nin);
  bfp_quant_kernel<<<(int)(nwt / 9216), 256, 0, stream>>>(wgt, qw, nwt);

  dim3 grid(GEMM_N / BN, GEMM_M / BM);  // (36, 32)
  bfp_gemm_wmma<<<grid, dim3(256), 0, stream>>>(qx, qw, bias, out);
}
